// SelfAttension_75539884802992
// MI455X (gfx1250) — compile-verified, matrix operand order checked by reading, not run on hardware
//
#include <hip/hip_runtime.h>
#include <hip/hip_bf16.h>

// ---------- constants ----------
#define BATCH 8
#define SEQ   2048
#define EMBED 768
#define HD    64
#define QKVN  192           // 3 * HD
#define ROWS  (BATCH * SEQ) // 16384
#define SCALE 0.125f        // 64^-0.5
#define LOG2E 1.44269504088896f

typedef _Float16 v4h  __attribute__((ext_vector_type(4)));
typedef _Float16 v8h  __attribute__((ext_vector_type(8)));
typedef _Float16 v16h __attribute__((ext_vector_type(16)));
typedef float    v8f  __attribute__((ext_vector_type(8)));

union H16 { v16h v; struct { v8h lo; v8h hi; } h; };

static __device__ __forceinline__ v8f wmma_f16(v16h a, v16h b, v8f c) {
    // D = A(16x32 f16) * B(32x16 f16) + C(16x16 f32)
    return __builtin_amdgcn_wmma_f32_16x16x32_f16(false, a, false, b, (short)0, c,
                                                  false, false);
}

// Load a 16x32 f16 A-operand (== B-operand of the transpose) from a row-major
// matrix with leading dimension ld (in halves), rows starting at row0, k-block kb.
static __device__ __forceinline__ v16h load_op(const _Float16* __restrict__ base,
                                               int row0, int ld, int kb,
                                               int lo, int hi) {
    const _Float16* r = base + (size_t)(row0 + lo) * ld + kb + hi * 8;
    H16 u;
    u.h.lo = *(const v8h*)(r);
    u.h.hi = *(const v8h*)(r + 16);
    return u.v;
}

// ---------- kernel 1a: x (f32) -> xh (f16) ----------
__global__ __launch_bounds__(256) void k_convert_x(const float* __restrict__ x,
                                                   _Float16* __restrict__ xh) {
    size_t i = ((size_t)blockIdx.x * 256 + threadIdx.x) * 4;
    float4 f = *(const float4*)(x + i);
    v4h h = { (_Float16)f.x, (_Float16)f.y, (_Float16)f.z, (_Float16)f.w };
    *(v4h*)(xh + i) = h;
}

// ---------- kernel 1b: w_qkv (768x192 f32) -> Wt (192x768 f16) ----------
__global__ __launch_bounds__(256) void k_tr_wqkv(const float* __restrict__ w,
                                                 _Float16* __restrict__ wt) {
    int idx = blockIdx.x * 256 + threadIdx.x;      // < 147456
    int e = idx / QKVN, n = idx % QKVN;
    wt[(size_t)n * EMBED + e] = (_Float16)w[idx];
}

// ---------- kernel 1c: w_proj (64x64 f32) -> Wpt (64x64 f16, transposed) ----------
__global__ __launch_bounds__(256) void k_tr_wproj(const float* __restrict__ w,
                                                  _Float16* __restrict__ wpt) {
    int idx = blockIdx.x * 256 + threadIdx.x;      // < 4096
    int k = idx / HD, n = idx % HD;
    wpt[(size_t)n * HD + k] = (_Float16)w[idx];
}

// ---------- kernel 2: QKV projection ----------
// One wave computes a full 16x192 row-block (12 resident accumulators).
// All 12 B operands are loaded into distinct registers BEFORE the WMMA chain so
// the scheduler can overlap loads with matrix ops (partial loadcnt waits)
// instead of draining the load pipe before every WMMA.
__global__ __launch_bounds__(128) void k_qkv(const _Float16* __restrict__ xh,
                                             const _Float16* __restrict__ wt,
                                             _Float16* __restrict__ qh,
                                             _Float16* __restrict__ kh,
                                             _Float16* __restrict__ vt) {
    const int lane = threadIdx.x & 31;
    const int lo = lane & 15, hi = lane >> 4;
    const int mt = blockIdx.x * 4 + (threadIdx.x >> 5);  // < 1024
    const int m0 = mt * 16;

    const _Float16* xrow = xh + (size_t)(m0 + lo) * EMBED + hi * 8;

    v8f acc[12];
#pragma unroll
    for (int nt = 0; nt < 12; ++nt) acc[nt] = (v8f){};

#pragma unroll 1
    for (int kb = 0; kb < EMBED; kb += 32) {
        H16 a;
        a.h.lo = *(const v8h*)(xrow + kb);
        a.h.hi = *(const v8h*)(xrow + kb + 16);

        v16h bv[12];
#pragma unroll
        for (int nt = 0; nt < 12; ++nt)
            bv[nt] = load_op(wt, nt * 16, EMBED, kb, lo, hi);

        if (kb + 32 < EMBED)  // next A k-block (x is the HBM-resident stream)
            __builtin_prefetch(xrow + kb + 32, 0, 0);

#pragma unroll
        for (int nt = 0; nt < 12; ++nt)
            acc[nt] = wmma_f16(a.v, bv[nt], acc[nt]);
    }

    const int bidx = m0 >> 11;
    const int rin = m0 & (SEQ - 1);
#pragma unroll
    for (int nt = 0; nt < 12; ++nt) {
#pragma unroll
        for (int r = 0; r < 8; ++r) {
            int row = m0 + r + 8 * hi;
            _Float16 v = (_Float16)acc[nt][r];
            if (nt < 4) {
                qh[(size_t)row * HD + nt * 16 + lo] = v;
            } else if (nt < 8) {
                kh[(size_t)row * HD + (nt - 4) * 16 + lo] = v;
            } else {
                vt[(size_t)bidx * HD * SEQ +
                   (size_t)((nt - 8) * 16 + lo) * SEQ + (rin + r + 8 * hi)] = v;
            }
        }
    }
}

// ---------- kernel 3: flash attention + output projection ----------
// One wave owns 32 query rows (two 16-row tiles): K/V operands shared across
// both tiles -> 16 WMMAs per 8 global b128 operand loads per 32-key step.
__global__ __launch_bounds__(128) void k_attn(const _Float16* __restrict__ qh,
                                              const _Float16* __restrict__ kh,
                                              const _Float16* __restrict__ vt,
                                              const _Float16* __restrict__ wpt,
                                              const float* __restrict__ bias,
                                              float* __restrict__ out) {
    __shared__ __align__(16) _Float16 smem[4][32 * 64];
    const int lane = threadIdx.x & 31;
    const int lo = lane & 15, hi = lane >> 4;
    const int wave = threadIdx.x >> 5;
    const int gw = blockIdx.x * 4 + wave;   // < 512
    const int m0 = gw * 32;                 // first of 32 query rows
    const int b = m0 >> 11;
    _Float16* pbuf = smem[wave];

    // resident Q operands: two row-tiles x two k-blocks
    v16h qa[2][2];
#pragma unroll
    for (int t = 0; t < 2; ++t) {
        qa[t][0] = load_op(qh, m0 + t * 16, HD, 0, lo, hi);
        qa[t][1] = load_op(qh, m0 + t * 16, HD, 32, lo, hi);
    }

    const _Float16* kbase = kh + (size_t)b * SEQ * HD;
    const _Float16* vbase = vt + (size_t)b * HD * SEQ;

    v8f o[2][4];
    float mr[2][8], ls[2][8];
#pragma unroll
    for (int t = 0; t < 2; ++t) {
#pragma unroll
        for (int n = 0; n < 4; ++n) o[t][n] = (v8f){};
#pragma unroll
        for (int r = 0; r < 8; ++r) { mr[t][r] = -3.0e38f; ls[t][r] = 0.0f; }
    }

    for (int kv = 0; kv < SEQ; kv += 32) {
        // shared K operands for S, shared V operands for PV (distinct registers)
        const v16h kb0 = load_op(kbase, kv,      HD, 0,  lo, hi);
        const v16h kb1 = load_op(kbase, kv,      HD, 32, lo, hi);
        const v16h kb2 = load_op(kbase, kv + 16, HD, 0,  lo, hi);
        const v16h kb3 = load_op(kbase, kv + 16, HD, 32, lo, hi);
        const v16h vb0 = load_op(vbase,  0, SEQ, kv, lo, hi);
        const v16h vb1 = load_op(vbase, 16, SEQ, kv, lo, hi);
        const v16h vb2 = load_op(vbase, 32, SEQ, kv, lo, hi);
        const v16h vb3 = load_op(vbase, 48, SEQ, kv, lo, hi);

        if (kv + 32 < SEQ) {  // prefetch next K/V tiles
            __builtin_prefetch(kbase + (size_t)(kv + 32 + lane) * HD, 0, 0);
            __builtin_prefetch(vbase + (size_t)lane * SEQ + kv + 32, 0, 0);
            __builtin_prefetch(vbase + (size_t)(32 + lane) * SEQ + kv + 32, 0, 0);
        }

#pragma unroll
        for (int t = 0; t < 2; ++t) {
            // ---- S = Q @ K^T (16 rows x 32 keys) ----
            v8f s0 = {}, s1 = {};
            s0 = wmma_f16(qa[t][0], kb0, s0);
            s0 = wmma_f16(qa[t][1], kb1, s0);
            s1 = wmma_f16(qa[t][0], kb2, s1);
            s1 = wmma_f16(qa[t][1], kb3, s1);

            // ---- online softmax: row M = r + 8*hi lives in one 16-lane half ----
#pragma unroll
            for (int r = 0; r < 8; ++r) {
                float a0 = s0[r] * SCALE;
                float a1 = s1[r] * SCALE;
                float t0 = fmaxf(a0, a1);
                t0 = fmaxf(t0, __shfl_xor(t0, 1, 32));
                t0 = fmaxf(t0, __shfl_xor(t0, 2, 32));
                t0 = fmaxf(t0, __shfl_xor(t0, 4, 32));
                t0 = fmaxf(t0, __shfl_xor(t0, 8, 32));
                float mnew = fmaxf(mr[t][r], t0);
                float alpha = exp2f((mr[t][r] - mnew) * LOG2E);
                float p0 = exp2f((a0 - mnew) * LOG2E);
                float p1 = exp2f((a1 - mnew) * LOG2E);
                float rs = p0 + p1;
                rs += __shfl_xor(rs, 1, 32);
                rs += __shfl_xor(rs, 2, 32);
                rs += __shfl_xor(rs, 4, 32);
                rs += __shfl_xor(rs, 8, 32);
                ls[t][r] = ls[t][r] * alpha + rs;
                mr[t][r] = mnew;
                o[t][0][r] *= alpha; o[t][1][r] *= alpha;
                o[t][2][r] *= alpha; o[t][3][r] *= alpha;
                // stage P tile (rows t*16.., 32 wide) in this wave's LDS slice
                pbuf[(t * 16 + r + 8 * hi) * 32 + lo]      = (_Float16)p0;
                pbuf[(t * 16 + r + 8 * hi) * 32 + 16 + lo] = (_Float16)p1;
            }
        }

        // ---- O += P @ V for both row tiles (V operands reused) ----
#pragma unroll
        for (int t = 0; t < 2; ++t) {
            H16 pa;
            const _Float16* pr = pbuf + (t * 16 + lo) * 32 + hi * 8;
            pa.h.lo = *(const v8h*)(pr);
            pa.h.hi = *(const v8h*)(pr + 16);
            o[t][0] = wmma_f16(pa.v, vb0, o[t][0]);
            o[t][1] = wmma_f16(pa.v, vb1, o[t][1]);
            o[t][2] = wmma_f16(pa.v, vb2, o[t][2]);
            o[t][3] = wmma_f16(pa.v, vb3, o[t][3]);
        }
    }

    // ---- normalize and stage O (32x64 f16) in LDS ----
#pragma unroll
    for (int t = 0; t < 2; ++t) {
#pragma unroll
        for (int r = 0; r < 8; ++r) {
            float inv = 1.0f / ls[t][r];
            _Float16* orow = pbuf + (t * 16 + r + 8 * hi) * 64;
            orow[ 0 + lo] = (_Float16)(o[t][0][r] * inv);
            orow[16 + lo] = (_Float16)(o[t][1][r] * inv);
            orow[32 + lo] = (_Float16)(o[t][2][r] * inv);
            orow[48 + lo] = (_Float16)(o[t][3][r] * inv);
        }
    }

    // ---- fused projection: out = O @ w_proj + b_proj ----
#pragma unroll
    for (int t = 0; t < 2; ++t) {
        H16 oa0, oa1;
        const _Float16* orow = pbuf + (t * 16 + lo) * 64 + hi * 8;
        oa0.h.lo = *(const v8h*)(orow);
        oa0.h.hi = *(const v8h*)(orow + 16);
        oa1.h.lo = *(const v8h*)(orow + 32);
        oa1.h.hi = *(const v8h*)(orow + 48);
#pragma unroll
        for (int ct = 0; ct < 4; ++ct) {
            float bv = bias[ct * 16 + lo];
            v8f acc;
#pragma unroll
            for (int r = 0; r < 8; ++r) acc[r] = bv;
            acc = wmma_f16(oa0.v, load_op(wpt, ct * 16, HD, 0,  lo, hi), acc);
            acc = wmma_f16(oa1.v, load_op(wpt, ct * 16, HD, 32, lo, hi), acc);
#pragma unroll
            for (int r = 0; r < 8; ++r)
                out[(size_t)(m0 + t * 16 + r + 8 * hi) * HD + ct * 16 + lo] = acc[r];
        }
    }
}

// ---------- host launcher ----------
extern "C" void kernel_launch(void* const* d_in, const int* in_sizes, int n_in,
                              void* d_out, int out_size, void* d_ws, size_t ws_size,
                              hipStream_t stream) {
    (void)in_sizes; (void)n_in; (void)out_size; (void)ws_size;
    const float* x     = (const float*)d_in[0];
    const float* wqkv  = (const float*)d_in[1];
    const float* wproj = (const float*)d_in[2];
    const float* bproj = (const float*)d_in[3];
    float* out = (float*)d_out;

    char* ws = (char*)d_ws;
    _Float16* xh  = (_Float16*)(ws);                // 25,165,824 B
    _Float16* wt  = (_Float16*)(ws + 25165824);     //    294,912 B
    _Float16* wpt = (_Float16*)(ws + 25460736);     //      8,192 B
    _Float16* qh  = (_Float16*)(ws + 25468928);     //  2,097,152 B
    _Float16* kh  = (_Float16*)(ws + 27566080);     //  2,097,152 B
    _Float16* vt  = (_Float16*)(ws + 29663232);     //  2,097,152 B  (total ~30.3 MB)

    k_convert_x<<<ROWS * EMBED / 1024, 256, 0, stream>>>(x, xh);
    k_tr_wqkv<<<EMBED * QKVN / 256, 256, 0, stream>>>(wqkv, wt);
    k_tr_wproj<<<HD * HD / 256, 256, 0, stream>>>(wproj, wpt);
    k_qkv<<<(ROWS / 16) / 4, 128, 0, stream>>>(xh, wt, qh, kh, vt);
    k_attn<<<(ROWS / 32) / 4, 128, 0, stream>>>(qh, kh, vt, wpt, bproj, out);
}
